// SparseSelfAttention_10307921511046
// MI455X (gfx1250) — matrix operand order repaired and checked by reading, not compiled, not run on hardware
//
#include <hip/hip_runtime.h>
#include <stdint.h>

// Problem constants (match reference)
#define Bv   4
#define Hv   8
#define Sv   1024
#define Dv   64
#define BLK  64
#define NBv  16

typedef __attribute__((ext_vector_type(16))) __bf16 v16bf;
typedef __attribute__((ext_vector_type(8)))  float  v8f;

union Frag {
  v16bf    v;
  uint32_t u[8];
};

// Native RNE fp32->bf16 (lets the backend use V_CVT_PK_BF16_F32-class ops)
__device__ __forceinline__ unsigned short f2bf(float f) {
  return __builtin_bit_cast(unsigned short, (__bf16)f);
}
__device__ __forceinline__ uint32_t pkbf(float x, float y) {
  return (uint32_t)f2bf(x) | ((uint32_t)f2bf(y) << 16);
}

__global__ __launch_bounds__(128)
void SparseSelfAttention_wmma_kernel(const float* __restrict__ Q,
                                     const float* __restrict__ K,
                                     const float* __restrict__ V,
                                     const int*   __restrict__ layout,
                                     float*       __restrict__ O)
{
  const int qb   = blockIdx.x;   // query block
  const int h    = blockIdx.y;
  const int b    = blockIdx.z;
  const int tid  = threadIdx.x;
  const int wv   = tid >> 5;     // wave 0..3, owns 16 query rows
  const int lane = tid & 31;
  const int hlf  = lane >> 4;    // 0 or 1 (lane half)
  const int l16  = lane & 15;
  const int rbase = hlf * 8;     // C/D layout: lanes>=16 hold rows M=8..15

  // bf16 tiles in LDS; row stride 72 (16B-aligned, de-conflicted)
  __shared__ __align__(16) unsigned short Qs[64][72];          // [q][d]
  __shared__ __align__(16) unsigned short Ks[64][72];          // [k][d]
  __shared__ __align__(16) unsigned short Vt[64][72];          // [d][k] (transposed)
  __shared__ __align__(16) unsigned short Ps[4][16][72];       // per-wave P strip [q][k]

  const size_t headoff = ((size_t)(b * Hv + h)) * Sv * Dv;
  const float4* Qg4 = (const float4*)(Q + headoff + (size_t)qb * BLK * Dv);

  // fold softmax scale and log2(e) into Q so we can use exp2 throughout
  const float qscale = 0.125f * 1.44269504088896340736f;   // D^-0.5 * log2(e)

  // ---------------- load Q tile -> LDS (bf16, pre-scaled) ----------------
  #pragma unroll
  for (int i = 0; i < 8; ++i) {
    int c   = tid + i * 128;            // 1024 float4 chunks (64 rows x 16)
    int row = c >> 4;
    int col = (c & 15) * 4;
    float4 v = Qg4[c];
    uint2 pk;
    pk.x = pkbf(v.x * qscale, v.y * qscale);
    pk.y = pkbf(v.z * qscale, v.w * qscale);
    *(uint2*)&Qs[row][col] = pk;
  }
  __syncthreads();

  // ---------------- Q A-fragments (2 k-steps of 32 along D) --------------
  // 16-bit A layout: lane l16 = row; half 0 -> K {0..7,16..23}, half 1 -> +8
  Frag aQ[2];
  #pragma unroll
  for (int ks = 0; ks < 2; ++ks) {
    const int row = wv * 16 + l16;
    const int d0  = ks * 32 + hlf * 8;
    *(uint4*)&aQ[ks].u[0] = *(const uint4*)&Qs[row][d0];
    *(uint4*)&aQ[ks].u[4] = *(const uint4*)&Qs[row][d0 + 16];
  }

  float m_i[8], l_i[8];
  v8f oacc[4];
  #pragma unroll
  for (int r = 0; r < 8; ++r) { m_i[r] = -3.0e30f; l_i[r] = 0.0f; }
  #pragma unroll
  for (int nt = 0; nt < 4; ++nt)
    #pragma unroll
    for (int r = 0; r < 8; ++r) oacc[nt][r] = 0.0f;

  const int* lay = layout + ((size_t)h * NBv + qb) * NBv;

  for (int kb = 0; kb < NBv; ++kb) {
    if (lay[kb] == 0) continue;        // uniform branch: EXEC stays all-ones

    __syncthreads();                   // all waves done reading prev Ks/Vt

    const float4* Kg4 = (const float4*)(K + headoff + (size_t)kb * BLK * Dv);
    const float4* Vg4 = (const float4*)(V + headoff + (size_t)kb * BLK * Dv);
    #pragma unroll
    for (int i = 0; i < 8; ++i) {
      int c   = tid + i * 128;
      int row = c >> 4;
      int col = (c & 15) * 4;
      float4 kv = Kg4[c];
      uint2 pk;
      pk.x = pkbf(kv.x, kv.y);
      pk.y = pkbf(kv.z, kv.w);
      *(uint2*)&Ks[row][col] = pk;
      float4 vv = Vg4[c];                // store V transposed: Vt[d][k]
      Vt[col + 0][row] = f2bf(vv.x);
      Vt[col + 1][row] = f2bf(vv.y);
      Vt[col + 2][row] = f2bf(vv.z);
      Vt[col + 3][row] = f2bf(vv.w);
    }
    if (kb + 1 < NBv) {                // warm L2 for next key block
      __builtin_prefetch(K + headoff + (size_t)(kb + 1) * BLK * Dv + tid * 32, 0, 1);
      __builtin_prefetch(V + headoff + (size_t)(kb + 1) * BLK * Dv + tid * 32, 0, 1);
    }
    __syncthreads();

    // ---------------- S = Q K^T : 16x64 per wave (4 tiles x 2 k-steps) ---
    v8f sacc[4];
    #pragma unroll
    for (int nt = 0; nt < 4; ++nt) {
      v8f s;
      #pragma unroll
      for (int r = 0; r < 8; ++r) s[r] = 0.0f;
      #pragma unroll
      for (int ks = 0; ks < 2; ++ks) {
        // 16-bit B layout: lane l16 = column (key row); half 0 -> K {0..7,16..23}, half 1 -> +8
        Frag bK;
        const int krow = nt * 16 + l16;
        const int k0   = ks * 32 + hlf * 8;
        *(uint4*)&bK.u[0] = *(const uint4*)&Ks[krow][k0];
        *(uint4*)&bK.u[4] = *(const uint4*)&Ks[krow][k0 + 16];
        s = __builtin_amdgcn_wmma_f32_16x16x32_bf16(false, aQ[ks].v, false, bK.v,
                                                    (short)0, s, false, false);
      }
      sacc[nt] = s;
    }

    // ---------------- online softmax (exp2 domain) -----------------------
    float vmax[8];
    #pragma unroll
    for (int r = 0; r < 8; ++r) {
      float m = fmaxf(fmaxf(sacc[0][r], sacc[1][r]), fmaxf(sacc[2][r], sacc[3][r]));
      vmax[r] = m;
    }
    #pragma unroll
    for (int off = 1; off < 16; off <<= 1)
      #pragma unroll
      for (int r = 0; r < 8; ++r)
        vmax[r] = fmaxf(vmax[r], __shfl_xor(vmax[r], off));

    float alpha[8], lsum[8];
    #pragma unroll
    for (int r = 0; r < 8; ++r) {
      float mnew = fmaxf(m_i[r], vmax[r]);
      alpha[r] = exp2f(m_i[r] - mnew);
      m_i[r]   = mnew;
      lsum[r]  = 0.0f;
    }

    #pragma unroll
    for (int nt = 0; nt < 4; ++nt)
      #pragma unroll
      for (int r = 0; r < 8; ++r) {
        float p  = exp2f(sacc[nt][r] - m_i[r]);
        lsum[r] += p;
        Ps[wv][rbase + r][nt * 16 + l16] = f2bf(p);   // C-layout -> LDS
      }

    #pragma unroll
    for (int off = 1; off < 16; off <<= 1)
      #pragma unroll
      for (int r = 0; r < 8; ++r)
        lsum[r] += __shfl_xor(lsum[r], off);

    #pragma unroll
    for (int r = 0; r < 8; ++r) l_i[r] = l_i[r] * alpha[r] + lsum[r];
    #pragma unroll
    for (int nt = 0; nt < 4; ++nt)
      #pragma unroll
      for (int r = 0; r < 8; ++r) oacc[nt][r] *= alpha[r];

    // ---------------- O += P V -------------------------------------------
    Frag aP[2];
    #pragma unroll
    for (int ks = 0; ks < 2; ++ks) {
      const int k0 = ks * 32 + hlf * 8;
      *(uint4*)&aP[ks].u[0] = *(const uint4*)&Ps[wv][l16][k0];
      *(uint4*)&aP[ks].u[4] = *(const uint4*)&Ps[wv][l16][k0 + 16];
    }
    #pragma unroll
    for (int nt = 0; nt < 4; ++nt) {
      #pragma unroll
      for (int ks = 0; ks < 2; ++ks) {
        Frag bV;
        const int drow = nt * 16 + l16;          // output column d = Vt row
        const int k0   = ks * 32 + hlf * 8;
        *(uint4*)&bV.u[0] = *(const uint4*)&Vt[drow][k0];
        *(uint4*)&bV.u[4] = *(const uint4*)&Vt[drow][k0 + 16];
        oacc[nt] = __builtin_amdgcn_wmma_f32_16x16x32_bf16(false, aP[ks].v, false, bV.v,
                                                           (short)0, oacc[nt], false, false);
      }
    }
  }

  // ---------------- normalize + store -----------------------------------
  float* Og = O + headoff + (size_t)qb * BLK * Dv;
  float rl[8];
  #pragma unroll
  for (int r = 0; r < 8; ++r) rl[r] = 1.0f / l_i[r];
  #pragma unroll
  for (int nt = 0; nt < 4; ++nt)
    #pragma unroll
    for (int r = 0; r < 8; ++r) {
      const int m = rbase + r;
      Og[(size_t)(wv * 16 + m) * Dv + nt * 16 + l16] = oacc[nt][r] * rl[r];
    }
}

extern "C" void kernel_launch(void* const* d_in, const int* in_sizes, int n_in,
                              void* d_out, int out_size, void* d_ws, size_t ws_size,
                              hipStream_t stream) {
  (void)in_sizes; (void)n_in; (void)out_size; (void)d_ws; (void)ws_size;
  const float* Q      = (const float*)d_in[0];
  const float* K      = (const float*)d_in[1];
  const float* V      = (const float*)d_in[2];
  const int*   layout = (const int*)d_in[3];
  float*       O      = (float*)d_out;

  dim3 grid(NBv, Hv, Bv);   // one workgroup per (query-block, head, batch)
  dim3 block(128);          // 4 waves, 16 query rows each
  hipLaunchKernelGGL(SparseSelfAttention_wmma_kernel, grid, block, 0, stream,
                     Q, K, V, layout, O);
}
